// DGLJTNNEncoder_2379411882635
// MI455X (gfx1250) — matrix operand order repaired and checked by reading, not compiled, not run on hardware
//
#include <hip/hip_runtime.h>
#include <hip/hip_bf16.h>
#include <math.h>

// ---- problem constants (from reference) ----
#define NPT 127
#define NB  128
#define HH  450                 // hidden size
#define HP  512                 // padded hidden (N and half-K)
#define NNODES (NB*NPT)         // 16256
#define EUP (NB*(NPT-1))        // 16128
#define EE  (2*EUP)             // 32256
#define NT  32                  // N tiles  (HP/16)
#define KT  32                  // K tiles  (2*HP/32)

typedef __attribute__((ext_vector_type(16))) _Float16 v16h;
typedef __attribute__((ext_vector_type(8)))  _Float16 v8h;
typedef __attribute__((ext_vector_type(8)))  float    v8f;

__device__ __forceinline__ v8f wmma16(v16h a, v16h b, v8f c) {
  // D = A(16x32 f16) * B(32x16 f16) + C(16x16 f32)
  return __builtin_amdgcn_wmma_f32_16x16x32_f16(false, a, false, b, (short)0, c,
                                                false, false);
}

// A fragment from an f16 row (global or LDS; row padded to >= k0+32 halves).
// Layout: lane l holds row M=l&15; halves 0..7 -> K=k0+off+0..7,
// halves 8..15 -> K=k0+16+off+0..7, off = (l>=16)?8:0.
// Compiles to two 16-byte vector loads (global_load_b128 / ds_load_b128).
__device__ __forceinline__ v16h load_a_row(const _Float16* rowp, int k0, int lane) {
  int off = (lane >= 16) ? 8 : 0;
  v8h lo = *(const v8h*)(rowp + k0 + off);
  v8h hi = *(const v8h*)(rowp + k0 + 16 + off);
  v16h a;
#pragma unroll
  for (int i = 0; i < 8; ++i) { a[i] = lo[i]; a[8 + i] = hi[i]; }
  return a;
}

// B fragment from pre-swizzled weight buffer: [kt][nt][lane][16h] contiguous.
__device__ __forceinline__ v16h load_b_sw(const _Float16* W, int kt, int nt, int lane) {
  return *(const v16h*)(W + (size_t)((kt * NT + nt) * 32 + lane) * 16);
}

__device__ __forceinline__ float sigmf(float x) { return 1.0f / (1.0f + expf(-x)); }

// Incoming edges of a node in the perfect-binary-tree forest:
// parent->node down-edge (if not root) + the two child->node up-edges (if not leaf).
__device__ __forceinline__ int in_edges(int node, int* e) {
  int b  = node / NPT;
  int ls = node - b * NPT;
  int eb = b * (NPT - 1);
  int n = 0;
  if (ls > 0) e[n++] = EUP + eb + (ls - 1);
  int c1 = 2 * ls + 1;
  if (c1 < NPT) { e[n++] = eb + (c1 - 1); e[n++] = eb + c1; }
  return n;
}

// ---------------- setup kernels ----------------
__global__ void zeroF(float* __restrict__ p, size_t n) {
  for (size_t i = (size_t)blockIdx.x * blockDim.x + threadIdx.x; i < n;
       i += (size_t)gridDim.x * blockDim.x)
    p[i] = 0.0f;
}

__global__ void gatherX(const int* __restrict__ wid, const float* __restrict__ emb,
                        _Float16* __restrict__ xh) {
  int node = blockIdx.x;
  int w = wid[node];
  for (int n = threadIdx.x; n < HP; n += blockDim.x) {
    float v = (n < HH) ? emb[(size_t)w * HH + n] : 0.0f;
    xh[(size_t)node * HP + n] = (_Float16)v;
  }
}

// Build 4 f16 weight buffers in WMMA-B fragment order, padded K=1024, N=512:
// buf0: [Wz_top; 0; Wz_bot; 0]  buf1: [Wh...]  buf2: [Wr; 0; Ur; 0]  buf3: [Wg...]
__global__ void convW(const float* __restrict__ Wz, const float* __restrict__ Wh,
                      const float* __restrict__ Wr, const float* __restrict__ Ur,
                      const float* __restrict__ Wg, _Float16* __restrict__ out) {
  const size_t PER = (size_t)KT * NT * 32 * 16;   // 524288 per buffer
  for (size_t i = (size_t)blockIdx.x * blockDim.x + threadIdx.x; i < 4 * PER;
       i += (size_t)gridDim.x * blockDim.x) {
    int buf = (int)(i / PER);
    int rr  = (int)(i % PER);
    int h    = rr & 15;
    int lane = (rr >> 4) & 31;
    int nt   = (rr >> 9) & 31;
    int kt   = rr >> 14;
    int k = kt * 32 + (h & 7) + ((h >= 8) ? 16 : 0) + ((lane >= 16) ? 8 : 0);   // padded K index
    int n = nt * 16 + (lane & 15);                   // padded N index
    float val = 0.0f;
    if (n < HH) {
      if (buf == 0) {
        if (k < HH) val = Wz[(size_t)k * HH + n];
        else if (k >= HP && k < HP + HH) val = Wz[(size_t)(k - HP + HH) * HH + n];
      } else if (buf == 1) {
        if (k < HH) val = Wh[(size_t)k * HH + n];
        else if (k >= HP && k < HP + HH) val = Wh[(size_t)(k - HP + HH) * HH + n];
      } else if (buf == 2) {
        if (k < HH) val = Wr[(size_t)k * HH + n];
        else if (k >= HP && k < HP + HH) val = Ur[(size_t)(k - HP) * HH + n];
      } else {
        if (k < HH) val = Wg[(size_t)k * HH + n];
        else if (k >= HP && k < HP + HH) val = Wg[(size_t)(k - HP + HH) * HH + n];
      }
    }
    out[i] = (_Float16)val;
  }
}

// ---------------- per-level kernel A: z / mt / m_new ----------------
// Grid: (level_cnt/16, 4); block 256 = 8 waves; wave -> one 16-col tile.
__global__ void levelA(const int* __restrict__ sched, const int* __restrict__ src,
                       const int* __restrict__ rev,
                       float* __restrict__ m, const float* __restrict__ rm,
                       const _Float16* __restrict__ xh, _Float16* __restrict__ mnewh,
                       const _Float16* __restrict__ Wzs, const _Float16* __restrict__ Whs,
                       const float* __restrict__ bz, const float* __restrict__ bh,
                       int lvl, int Lmax) {
  __shared__ float    s32[16 * 464];    // f32 s, needed by gate epilogue
  __shared__ _Float16 sh16[16 * HP];    // f16 s, zero-padded, GEMM A operand
  __shared__ _Float16 ah16[16 * HP];    // f16 arm, zero-padded, GEMM A operand
  __shared__ int sh_eid[16];
  __shared__ int sh_se[16];
  int tid = threadIdx.x;
  if (tid < 16) {
    int idx = blockIdx.x * 16 + tid;
    int eid = (idx < Lmax) ? sched[lvl * Lmax + idx] : EE;
    sh_eid[tid] = eid;
    sh_se[tid]  = (eid < EE) ? src[eid] : 0;
  }
  __syncthreads();
  {   // gather s = sum(in m) - m[rev], arm = sum(in rm) - rm[rev] into LDS
    int row = tid >> 4, cl = tid & 15;
    int eid = sh_eid[row];
    int ine[3];
    int ni = 0, re = 0;
    if (eid < EE) { ni = in_edges(sh_se[row], ine); re = rev[eid]; }
    for (int n = cl; n < HP; n += 16) {
      float sv = 0.0f, av = 0.0f;
      if (eid < EE && n < HH) {
        for (int i = 0; i < ni; ++i) {
          sv += m[(size_t)ine[i] * HH + n];
          av += rm[(size_t)ine[i] * HH + n];
        }
        sv -= m[(size_t)re * HH + n];
        av -= rm[(size_t)re * HH + n];
      }
      if (n < HH) s32[row * 464 + n] = sv;
      sh16[row * HP + n] = (_Float16)sv;
      ah16[row * HP + n] = (_Float16)av;
    }
  }
  __syncthreads();

  int lane = tid & 31, wave = tid >> 5;
  int nt = blockIdx.y * 8 + wave;
  int arow = lane & 15;
  const _Float16* sxp = xh + (size_t)sh_se[arow] * HP;
  const _Float16* slp = &sh16[arow * HP];
  const _Float16* alp = &ah16[arow * HP];
  v8f accz = {};
  v8f acch = {};
#pragma unroll 4
  for (int kt = 0; kt < 16; ++kt) {          // K half 1: sx (shared A fragment)
    v16h a = load_a_row(sxp, kt * 32, lane);
    accz = wmma16(a, load_b_sw(Wzs, kt, nt, lane), accz);
    acch = wmma16(a, load_b_sw(Whs, kt, nt, lane), acch);
  }
#pragma unroll 4
  for (int kt = 16; kt < 32; ++kt) {         // K half 2: s (for z), arm (for mt)
    int k0 = (kt - 16) * 32;
    accz = wmma16(load_a_row(slp, k0, lane), load_b_sw(Wzs, kt, nt, lane), accz);
    acch = wmma16(load_a_row(alp, k0, lane), load_b_sw(Whs, kt, nt, lane), acch);
  }
  int col = nt * 16 + (lane & 15);
  int rbase = (lane >= 16) ? 8 : 0;
#pragma unroll
  for (int v = 0; v < 8; ++v) {
    int r = v + rbase;
    int eid = sh_eid[r];
    float mn = 0.0f;
    if (eid < EE && col < HH) {
      float z  = sigmf(accz[v] + bz[col]);
      float mt = tanhf(acch[v] + bh[col]);
      float sv = s32[r * 464 + col];
      mn = (1.0f - z) * sv + z * mt;
      m[(size_t)eid * HH + col] = mn;
    }
    mnewh[(size_t)(blockIdx.x * 16 + r) * HP + col] = (_Float16)mn;  // padded f16
  }
}

// ---------------- per-level kernel B: r = sigmoid([dx,m_new]@[Wr;Ur]+bur) ----------------
__global__ void levelB(const int* __restrict__ sched, const int* __restrict__ dst,
                       const float* __restrict__ m, float* __restrict__ rmw,
                       const _Float16* __restrict__ xh, const _Float16* __restrict__ mnewh,
                       const _Float16* __restrict__ Wrus, const float* __restrict__ bur,
                       int lvl, int Lmax) {
  __shared__ int sh_eid[16];
  __shared__ int sh_de[16];
  int tid = threadIdx.x;
  if (tid < 16) {
    int idx = blockIdx.x * 16 + tid;
    int eid = (idx < Lmax) ? sched[lvl * Lmax + idx] : EE;
    sh_eid[tid] = eid;
    sh_de[tid]  = (eid < EE) ? dst[eid] : 0;
  }
  __syncthreads();
  int lane = tid & 31, wave = tid >> 5;
  int nt = blockIdx.y * 8 + wave;
  int arow = lane & 15;
  const _Float16* dxp = xh + (size_t)sh_de[arow] * HP;
  const _Float16* mnp = mnewh + (size_t)(blockIdx.x * 16 + arow) * HP;
  v8f acc = {};
#pragma unroll 4
  for (int kt = 0; kt < 16; ++kt)
    acc = wmma16(load_a_row(dxp, kt * 32, lane), load_b_sw(Wrus, kt, nt, lane), acc);
#pragma unroll 4
  for (int kt = 16; kt < 32; ++kt)
    acc = wmma16(load_a_row(mnp, (kt - 16) * 32, lane), load_b_sw(Wrus, kt, nt, lane), acc);
  int col = nt * 16 + (lane & 15);
  int rbase = (lane >= 16) ? 8 : 0;
#pragma unroll
  for (int v = 0; v < 8; ++v) {
    int r = v + rbase;
    int eid = sh_eid[r];
    if (eid < EE && col < HH) {
      float rg = sigmf(acc[v] + bur[col]);
      float mn = m[(size_t)eid * HH + col];
      rmw[(size_t)eid * HH + col] = rg * mn;
    }
  }
}

// ---------------- final readout: h = relu([x, node_m] @ Wg + bg) ----------------
__global__ void finalK(const float* __restrict__ m, const _Float16* __restrict__ xh,
                       const _Float16* __restrict__ Wgs, const float* __restrict__ bg,
                       float* __restrict__ out) {
  __shared__ _Float16 nm16[16 * HP];   // f16 node_m, zero-padded
  int tid = threadIdx.x;
  int nodeBase = blockIdx.x * 16;
  {
    int row = tid >> 4, cl = tid & 15;
    int ine[3];
    int ni = in_edges(nodeBase + row, ine);
    for (int n = cl; n < HP; n += 16) {
      float sv = 0.0f;
      if (n < HH)
        for (int i = 0; i < ni; ++i) sv += m[(size_t)ine[i] * HH + n];
      nm16[row * HP + n] = (_Float16)sv;
    }
  }
  __syncthreads();
  int lane = tid & 31, wave = tid >> 5;
  int nt = blockIdx.y * 8 + wave;
  int arow = lane & 15;
  const _Float16* sxp = xh + (size_t)(nodeBase + arow) * HP;
  const _Float16* nmp = &nm16[arow * HP];
  v8f acc = {};
#pragma unroll 4
  for (int kt = 0; kt < 16; ++kt)
    acc = wmma16(load_a_row(sxp, kt * 32, lane), load_b_sw(Wgs, kt, nt, lane), acc);
#pragma unroll 4
  for (int kt = 16; kt < 32; ++kt)
    acc = wmma16(load_a_row(nmp, (kt - 16) * 32, lane), load_b_sw(Wgs, kt, nt, lane), acc);
  int col = nt * 16 + (lane & 15);
  int rbase = (lane >= 16) ? 8 : 0;
#pragma unroll
  for (int v = 0; v < 8; ++v) {
    int r = v + rbase;
    if (col < HH) {
      float h = acc[v] + bg[col];
      out[(size_t)(nodeBase + r) * HH + col] = fmaxf(h, 0.0f);
    }
  }
}

extern "C" void kernel_launch(void* const* d_in, const int* in_sizes, int n_in,
                              void* d_out, int out_size, void* d_ws, size_t ws_size,
                              hipStream_t stream) {
  const int*   wid   = (const int*)d_in[0];
  const int*   src   = (const int*)d_in[1];
  const int*   dst   = (const int*)d_in[2];
  const int*   rev   = (const int*)d_in[3];
  const int*   sched = (const int*)d_in[4];
  const float* emb   = (const float*)d_in[5];
  const float* Wz    = (const float*)d_in[6];
  const float* bz    = (const float*)d_in[7];
  const float* Wr    = (const float*)d_in[8];
  const float* Ur    = (const float*)d_in[9];
  const float* bur   = (const float*)d_in[10];
  const float* Wh    = (const float*)d_in[11];
  const float* bh    = (const float*)d_in[12];
  const float* Wg    = (const float*)d_in[13];
  const float* bg    = (const float*)d_in[14];
  const int Lmax = in_sizes[4] / 12;
  (void)n_in; (void)out_size; (void)ws_size;

  auto align256 = [](size_t x) { return (x + 255) & ~(size_t)255; };
  char* ws = (char*)d_ws;
  size_t off = 0;
  float* m  = (float*)(ws + off); off = align256(off + (size_t)(EE + 1) * HH * sizeof(float));
  float* rm = (float*)(ws + off); off = align256(off + (size_t)(EE + 1) * HH * sizeof(float));
  _Float16* xh    = (_Float16*)(ws + off); off = align256(off + (size_t)NNODES * HP * sizeof(_Float16));
  _Float16* mnewh = (_Float16*)(ws + off); off = align256(off + (size_t)8192 * HP * sizeof(_Float16));
  _Float16* wsw   = (_Float16*)(ws + off);
  const size_t PER = (size_t)KT * NT * 32 * 16;
  _Float16* Wz_sw  = wsw;
  _Float16* Wh_sw  = wsw + PER;
  _Float16* Wru_sw = wsw + 2 * PER;
  _Float16* Wg_sw  = wsw + 3 * PER;

  zeroF<<<4096, 256, 0, stream>>>(m,  (size_t)(EE + 1) * HH);
  zeroF<<<4096, 256, 0, stream>>>(rm, (size_t)(EE + 1) * HH);
  gatherX<<<NNODES, 256, 0, stream>>>(wid, emb, xh);
  convW<<<8192, 256, 0, stream>>>(Wz, Wh, Wr, Ur, Wg, wsw);

  static const int cnt[12] = {8192, 4096, 2048, 1024, 512, 256,
                              256, 512, 1024, 2048, 4096, 8192};
  for (int lvl = 0; lvl < 12; ++lvl) {
    dim3 g(cnt[lvl] / 16, 4);
    levelA<<<g, 256, 0, stream>>>(sched, src, rev, m, rm, xh, mnewh,
                                  Wz_sw, Wh_sw, bz, bh, lvl, Lmax);
    levelB<<<g, 256, 0, stream>>>(sched, dst, m, rm, xh, mnewh,
                                  Wru_sw, bur, lvl, Lmax);
  }
  finalK<<<dim3(NNODES / 16, 4), 256, 0, stream>>>(m, xh, Wg_sw, bg, (float*)d_out);
}
